// Decoder_12876311953617
// MI455X (gfx1250) — matrix operand order repaired and checked by reading, not compiled, not run on hardware
//
#include <hip/hip_runtime.h>
#include <math.h>

// ---------------- problem constants (from reference) ----------------
#define BB      32
#define SRC_LEN 128
#define TGT_LEN 64
#define DS      512          // decoder state
#define DWE     512          // embedding dim
#define H2      1024         // 2*D_H
#define DL2     1024         // 2*D_L (pre-maxout width)
#define VOCAB   32000
#define KX      2048         // concat input [s, we, c]
#define NG      2560         // concat gate outputs [t(1024), z(512), r(512), u(512)]

typedef unsigned short u16;
typedef __attribute__((ext_vector_type(16))) __bf16 v16bf;
typedef __attribute__((ext_vector_type(8)))  float  v8f;
typedef unsigned uint32x4 __attribute__((ext_vector_type(4)));

union Frag {
  u16      h[16];
  unsigned d[8];
  uint32x4 q[2];
  v16bf    v;
};

__device__ inline u16 f2bf(float f) {
  unsigned u = __float_as_uint(f);
  unsigned r = u + 0x7FFFu + ((u >> 16) & 1u);   // round-to-nearest-even
  return (u16)(r >> 16);
}

// A fragment: 16x32 bf16, A row-major [M][K] (u16 storage).
// lane L: row m = mBase + (L&15); k-half = (L>>4)*8
// VGPR v: k = ((v&4)?16:0) + khalf + (v&3)*2 (+1)
__device__ inline v16bf load_a_frag(const u16* __restrict__ A, int K,
                                    int mBase, int kBase, int lane) {
  Frag f;
  const u16* row = A + (size_t)(mBase + (lane & 15)) * K + kBase + ((lane >> 4) << 3);
  const unsigned* r32 = (const unsigned*)row;   // 4-byte aligned (even u16 offsets)
#pragma unroll
  for (int i = 0; i < 4; ++i) {
    f.d[i]     = r32[i];       // k = khalf + 2i, 2i+1
    f.d[i + 4] = r32[8 + i];   // k = khalf + 16 + 2i, 2i+1
  }
  return f.v;
}

// B fragment from packed layout: [kstep][ntile][lane][16 halves], 32B/lane contiguous
__device__ inline v16bf load_b_frag(const u16* __restrict__ Bp, int NT,
                                    int kstep, int ntile, int lane) {
  const uint32x4* q =
      (const uint32x4*)(Bp + ((((size_t)kstep * NT + ntile) << 5) + lane) * 16);
  Frag f;
  f.q[0] = q[0];
  f.q[1] = q[1];
  return f.v;
}

// ---------------- weight packing: fp32 [Ksub][Nsub] -> bf16 B fragments ----------------
// dst layout covers the full [Ktot][Ntot] packed tensor; this writes the tile-aligned
// sub-block at (k0, n0). W == nullptr writes zeros (used for the u-path s-rows).
__global__ void pack_b_kernel(const float* __restrict__ W, u16* __restrict__ dst,
                              int Ksub, int Nsub, int k0, int n0, int NT) {
  int idx = blockIdx.x * blockDim.x + threadIdx.x;
  int ntl_cnt = Nsub >> 4;
  int total = (Ksub >> 5) * ntl_cnt * 512;
  if (idx >= total) return;
  int j    = idx & 15;
  int lane = (idx >> 4) & 31;
  int tt   = idx >> 9;
  int ntl  = tt % ntl_cnt;
  int kst  = tt / ntl_cnt;
  int k_in = (kst << 5) + ((lane >> 4) << 3) + (j & 7) + ((j >> 3) << 4);   // B frag: lanes 0-15 K=0..7,16..23; 16-31 K=8..15,24..31
  int n_in = (ntl << 4) + (lane & 15);
  float v  = W ? W[(size_t)k_in * Nsub + n_in] : 0.0f;
  int kstep = (k0 >> 5) + kst;
  int ntile = (n0 >> 4) + ntl;
  dst[((((size_t)kstep * NT + ntile) << 5) + lane) * 16 + j] = f2bf(v);
}

__global__ void f32_to_bf16_kernel(const float* __restrict__ src, u16* __restrict__ dst, int n) {
  int i = blockIdx.x * blockDim.x + threadIdx.x;
  if (i < n) dst[i] = f2bf(src[i]);
}

// ---------------- generic one-tile-per-wave WMMA GEMM (used for h_proj) ----------------
__global__ void wmma_gemm_tile(const u16* __restrict__ A, int K,
                               const u16* __restrict__ Bp,
                               const float* __restrict__ bias,
                               float* __restrict__ C, int ldc,
                               int mtiles, int ntiles) {
  int lane = threadIdx.x & 31;
  int wave = blockIdx.x * (blockDim.x >> 5) + (threadIdx.x >> 5);
  if (wave >= mtiles * ntiles) return;          // wave-uniform: EXEC stays all-ones for WMMA
  int nt = wave % ntiles, mt = wave / ntiles;
  v8f acc = {};
  int ksteps = K >> 5;
  for (int ks = 0; ks < ksteps; ++ks) {
    v16bf a = load_a_frag(A, K, mt << 4, ks << 5, lane);
    v16bf b = load_b_frag(Bp, ntiles, ks, nt, lane);
    acc = __builtin_amdgcn_wmma_f32_16x16x32_bf16(false, a, false, b, (short)0, acc,
                                                  false, false);
  }
  int n  = (nt << 4) + (lane & 15);
  int mo = (mt << 4) + ((lane >> 4) << 3);
  float bs = bias ? bias[n] : 0.0f;
#pragma unroll
  for (int v = 0; v < 8; ++v) C[(size_t)(mo + v) * ldc + n] = acc[v] + bs;
}

// ---------------- M=32 WMMA GEMM: one N-tile per wave, both M-tiles share the B frag ----
__global__ void wmma_gemm_m32(const u16* __restrict__ A /*[32][K]*/, int K,
                              const u16* __restrict__ Bp,
                              const float* __restrict__ bias,
                              float* __restrict__ C, unsigned long long ldc,
                              int ntiles) {
  int lane = threadIdx.x & 31;
  int wave = blockIdx.x * (blockDim.x >> 5) + (threadIdx.x >> 5);
  if (wave >= ntiles) return;
  int nt = wave;
  v8f acc0 = {}, acc1 = {};
  int ksteps = K >> 5;
  for (int ks = 0; ks < ksteps; ++ks) {
    v16bf b  = load_b_frag(Bp, ntiles, ks, nt, lane);
    v16bf a0 = load_a_frag(A, K, 0,  ks << 5, lane);
    v16bf a1 = load_a_frag(A, K, 16, ks << 5, lane);
    acc0 = __builtin_amdgcn_wmma_f32_16x16x32_bf16(false, a0, false, b, (short)0, acc0,
                                                   false, false);
    acc1 = __builtin_amdgcn_wmma_f32_16x16x32_bf16(false, a1, false, b, (short)0, acc1,
                                                   false, false);
  }
  int n  = (nt << 4) + (lane & 15);
  int mo = (lane >> 4) << 3;
  float bs = bias ? bias[n] : 0.0f;
#pragma unroll
  for (int v = 0; v < 8; ++v) {
    C[(size_t)(mo + v) * ldc + n]      = acc0[v] + bs;
    C[(size_t)(16 + mo + v) * ldc + n] = acc1[v] + bs;
  }
}

// ---------------- attention: e = av . tanh(sp + hproj), softmax, c = attn @ hidden ------
__global__ void attention_kernel(const float* __restrict__ sp,      // [B][DS]
                                 const float* __restrict__ hproj,   // [B*L][DS]
                                 const float* __restrict__ hidden,  // [B*L][H2]
                                 const float* __restrict__ avW,     // [DS]
                                 const float* __restrict__ avb,     // [1]
                                 float* __restrict__ cbuf) {        // [B][H2]
  const int b = blockIdx.x;
  const int tid = threadIdx.x;          // 128 threads == SRC_LEN
  __shared__ float attn[SRC_LEN];
  __shared__ float red[SRC_LEN];

  const float* srow = sp + b * DS;
  const float* hp   = hproj + ((size_t)b * SRC_LEN + tid) * DS;
  float e = 0.0f;
  for (int k = 0; k < DS; ++k) e += tanhf(srow[k] + hp[k]) * avW[k];
  e += avb[0];

  red[tid] = e; __syncthreads();
  for (int o = SRC_LEN / 2; o > 0; o >>= 1) {
    if (tid < o) red[tid] = fmaxf(red[tid], red[tid + o]);
    __syncthreads();
  }
  float mx = red[0]; __syncthreads();
  float ex = __expf(e - mx);
  attn[tid] = ex; red[tid] = ex; __syncthreads();
  for (int o = SRC_LEN / 2; o > 0; o >>= 1) {
    if (tid < o) red[tid] += red[tid + o];
    __syncthreads();
  }
  float inv = 1.0f / red[0];
  attn[tid] *= inv; __syncthreads();

  const float* hb = hidden + (size_t)b * SRC_LEN * H2;
  for (int o = tid; o < H2; o += SRC_LEN) {
    float acc = 0.0f;
    for (int l = 0; l < SRC_LEN; ++l) acc += attn[l] * hb[(size_t)l * H2 + o];
    cbuf[b * H2 + o] = acc;
  }
}

// ---------------- build x = [s, E[y], c] in bf16 ----------------
__global__ void build_x_kernel(const float* __restrict__ s, const float* __restrict__ E,
                               const int* __restrict__ y, const float* __restrict__ cbuf,
                               u16* __restrict__ x) {
  int idx = blockIdx.x * blockDim.x + threadIdx.x;
  if (idx >= BB * KX) return;
  int b = idx >> 11, j = idx & (KX - 1);
  float v;
  if (j < DS)            v = s[b * DS + j];
  else if (j < DS + DWE) v = E[(size_t)y[b] * DWE + (j - DS)];
  else                   v = cbuf[b * H2 + (j - DS - DWE)];
  x[idx] = f2bf(v);
}

// ---------------- concat gate biases ----------------
__global__ void build_gbias_kernel(const float* tsb, const float* tyb, const float* tcb,
                                   const float* zsb, const float* zyb, const float* zcb,
                                   const float* ryb, const float* rsb,
                                   const float* syb, const float* scb,
                                   float* __restrict__ gb) {
  int j = blockIdx.x * blockDim.x + threadIdx.x;
  if (j >= NG) return;
  if (j < 1024)       gb[j] = tsb[j] + tyb[j] + tcb[j];
  else if (j < 1536) { int k = j - 1024; gb[j] = zsb[k] + zyb[k] + zcb[k]; }
  else if (j < 2048) { int k = j - 1536; gb[j] = rsb[k] + ryb[k] + zcb[k]; } // r reuses zc
  else               { int k = j - 2048; gb[j] = syb[k] + scb[k]; }
}

// ---------------- gates phase 2: sigmoid z/r, r*s, maxout t ----------------
__global__ void phase2_kernel(const float* __restrict__ G,   // [B][NG]
                              const float* __restrict__ s,   // [B][DS]
                              u16* __restrict__ rs_bf, u16* __restrict__ t_bf,
                              float* __restrict__ zbuf) {
  int idx = blockIdx.x * blockDim.x + threadIdx.x;
  if (idx >= BB * DS) return;
  int b = idx >> 9, j = idx & (DS - 1);
  const float* g = G + (size_t)b * NG;
  float z = 1.0f / (1.0f + __expf(-g[1024 + j]));
  float r = 1.0f / (1.0f + __expf(-g[1536 + j]));
  zbuf[idx]  = z;
  rs_bf[idx] = f2bf(r * s[idx]);
  t_bf[idx]  = f2bf(fmaxf(g[2 * j], g[2 * j + 1]));   // MaxPool1d(2) over adjacent pairs
}

// ---------------- state update: s = (1-z)s + z tanh(u + ss(r*s)) ----------------
__global__ void s_update_kernel(const float* __restrict__ G, const float* __restrict__ vbuf,
                                const float* __restrict__ zbuf,
                                float* __restrict__ s, u16* __restrict__ sbf) {
  int idx = blockIdx.x * blockDim.x + threadIdx.x;
  if (idx >= BB * DS) return;
  int b = idx >> 9, j = idx & (DS - 1);
  float u  = G[(size_t)b * NG + 2048 + j];
  float st = tanhf(u + vbuf[idx]);
  float z  = zbuf[idx];
  float sn = (1.0f - z) * s[idx] + z * st;
  s[idx]  = sn;
  sbf[idx] = f2bf(sn);
}

// ---------------- greedy argmax over vocab ----------------
__global__ void argmax_kernel(const float* __restrict__ logits, unsigned long long ldl,
                              int* __restrict__ y) {
  int b = blockIdx.x, tid = threadIdx.x;    // 256 threads
  const float* row = logits + (size_t)b * ldl;
  float best = -3.4e38f; int bi = 0;
  for (int v = tid; v < VOCAB; v += 256) {
    float x = row[v];
    if (x > best) { best = x; bi = v; }
  }
  __shared__ float bv[256];
  __shared__ int   bx[256];
  bv[tid] = best; bx[tid] = bi; __syncthreads();
  for (int o = 128; o > 0; o >>= 1) {
    if (tid < o) {
      if (bv[tid + o] > bv[tid] || (bv[tid + o] == bv[tid] && bx[tid + o] < bx[tid])) {
        bv[tid] = bv[tid + o]; bx[tid] = bx[tid + o];
      }
    }
    __syncthreads();
  }
  if (tid == 0) y[b] = bx[0];
}

__global__ void init_state_kernel(float* __restrict__ s, u16* __restrict__ sbf,
                                  int* __restrict__ y) {
  int i = blockIdx.x * blockDim.x + threadIdx.x;
  if (i < BB * DS) { s[i] = 0.0f; sbf[i] = 0; }
  if (i < BB) y[i] = 0;
}

// =====================================================================
extern "C" void kernel_launch(void* const* d_in, const int* in_sizes, int n_in,
                              void* d_out, int out_size, void* d_ws, size_t ws_size,
                              hipStream_t stream) {
  (void)in_sizes; (void)n_in; (void)out_size; (void)ws_size;
  const float* hidden = (const float*)d_in[0];
  const float* asW = (const float*)d_in[3],  *asb = (const float*)d_in[4];
  const float* ahW = (const float*)d_in[5],  *ahb = (const float*)d_in[6];
  const float* avW = (const float*)d_in[7],  *avb = (const float*)d_in[8];
  const float* tsW = (const float*)d_in[9],  *tsb = (const float*)d_in[10];
  const float* tyW = (const float*)d_in[11], *tyb = (const float*)d_in[12];
  const float* tcW = (const float*)d_in[13], *tcb = (const float*)d_in[14];
  const float* outW= (const float*)d_in[15], *outb= (const float*)d_in[16];
  const float* zyW = (const float*)d_in[17], *zyb = (const float*)d_in[18];
  const float* zsW = (const float*)d_in[19], *zsb = (const float*)d_in[20];
  const float* zcW = (const float*)d_in[21], *zcb = (const float*)d_in[22];
  const float* ryW = (const float*)d_in[23], *ryb = (const float*)d_in[24];
  const float* rsW = (const float*)d_in[25], *rsb = (const float*)d_in[26];
  const float* syW = (const float*)d_in[27], *syb = (const float*)d_in[28];
  const float* ssW = (const float*)d_in[29], *ssb = (const float*)d_in[30];
  const float* scW = (const float*)d_in[31], *scb = (const float*)d_in[32];
  const float* embed = (const float*)d_in[33];
  float* out = (float*)d_out;

  char* ws = (char*)d_ws;
  size_t off = 0;
  auto alloc = [&](size_t bytes) -> char* {
    char* p = ws + off;
    off = (off + bytes + 255) & ~(size_t)255;
    return p;
  };

  u16*   WoutP   = (u16*)alloc((size_t)(DS/32) * (VOCAB/16) * 512 * 2);   // 32.8 MB
  u16*   WgatesP = (u16*)alloc((size_t)(KX/32) * (NG/16) * 512 * 2);      // 10.5 MB
  u16*   WasP    = (u16*)alloc((size_t)(DS/32) * (DS/16) * 512 * 2);
  u16*   WssP    = (u16*)alloc((size_t)(DS/32) * (DS/16) * 512 * 2);
  u16*   WahP    = (u16*)alloc((size_t)(H2/32) * (DS/16) * 512 * 2);
  u16*   hidBF   = (u16*)alloc((size_t)BB * SRC_LEN * H2 * 2);            // 8.4 MB
  float* hproj   = (float*)alloc((size_t)BB * SRC_LEN * DS * 4);          // 8.4 MB
  float* sp      = (float*)alloc((size_t)BB * DS * 4);
  float* cbuf    = (float*)alloc((size_t)BB * H2 * 4);
  u16*   xbf     = (u16*)alloc((size_t)BB * KX * 2);
  float* G       = (float*)alloc((size_t)BB * NG * 4);
  float* gbias   = (float*)alloc((size_t)NG * 4);
  u16*   rs_bf   = (u16*)alloc((size_t)BB * DS * 2);
  u16*   t_bf    = (u16*)alloc((size_t)BB * DS * 2);
  float* zbuf    = (float*)alloc((size_t)BB * DS * 4);
  float* vbuf    = (float*)alloc((size_t)BB * DS * 4);
  float* sbuf    = (float*)alloc((size_t)BB * DS * 4);
  u16*   sbf     = (u16*)alloc((size_t)BB * DS * 2);
  int*   ybuf    = (int*)alloc((size_t)BB * 4);

  auto packB = [&](const float* W, u16* dst, int Ksub, int Nsub, int k0, int n0, int NT) {
    int total = (Ksub >> 5) * (Nsub >> 4) * 512;
    pack_b_kernel<<<(total + 255) / 256, 256, 0, stream>>>(W, dst, Ksub, Nsub, k0, n0, NT);
  };

  // ---- one-time per call: convert + pack weights into WMMA fragment layout ----
  packB(outW, WoutP, DS, VOCAB, 0, 0, VOCAB / 16);
  packB(asW,  WasP,  DS, DS,    0, 0, DS / 16);
  packB(ssW,  WssP,  DS, DS,    0, 0, DS / 16);
  packB(ahW,  WahP,  H2, DS,    0, 0, DS / 16);
  // concat gate weights: rows [s(0-511), we(512-1023), c(1024-2047)]
  packB(tsW,  WgatesP, DS, DL2, 0,    0,    NG / 16);
  packB(tyW,  WgatesP, DS, DL2, 512,  0,    NG / 16);
  packB(tcW,  WgatesP, H2, DL2, 1024, 0,    NG / 16);
  packB(zsW,  WgatesP, DS, DS,  0,    1024, NG / 16);
  packB(zyW,  WgatesP, DS, DS,  512,  1024, NG / 16);
  packB(zcW,  WgatesP, H2, DS,  1024, 1024, NG / 16);
  packB(rsW,  WgatesP, DS, DS,  0,    1536, NG / 16);
  packB(ryW,  WgatesP, DS, DS,  512,  1536, NG / 16);
  packB(zcW,  WgatesP, H2, DS,  1024, 1536, NG / 16);  // r-gate context reuses zc (reference)
  packB(nullptr, WgatesP, DS, DS, 0,  2048, NG / 16);  // u-path has no s term -> zeros
  packB(syW,  WgatesP, DS, DS,  512,  2048, NG / 16);
  packB(scW,  WgatesP, H2, DS,  1024, 2048, NG / 16);

  {
    int n = BB * SRC_LEN * H2;
    f32_to_bf16_kernel<<<(n + 255) / 256, 256, 0, stream>>>(hidden, hidBF, n);
  }
  build_gbias_kernel<<<(NG + 255) / 256, 256, 0, stream>>>(tsb, tyb, tcb, zsb, zyb, zcb,
                                                           ryb, rsb, syb, scb, gbias);

  // ---- hoisted: h_proj = hidden @ W_ah + b_ah  (M=4096, K=1024, N=512) ----
  {
    int mtiles = (BB * SRC_LEN) / 16, ntiles = DS / 16;   // 256 x 32 = 8192 waves
    int waves = mtiles * ntiles;
    wmma_gemm_tile<<<(waves * 32 + 255) / 256, 256, 0, stream>>>(
        hidBF, H2, WahP, ahb, hproj, DS, mtiles, ntiles);
  }

  init_state_kernel<<<(BB * DS + 255) / 256, 256, 0, stream>>>(sbuf, sbf, ybuf);

  // ---- 64 sequential decode steps ----
  for (int t = 0; t < TGT_LEN; ++t) {
    // sp = as_(s_prev)                (M=32, K=512, N=512)
    wmma_gemm_m32<<<(32 * 32 + 255) / 256, 256, 0, stream>>>(
        sbf, DS, WasP, asb, sp, (unsigned long long)DS, DS / 16);
    // additive attention + context
    attention_kernel<<<BB, SRC_LEN, 0, stream>>>(sp, hproj, hidden, avW, avb, cbuf);
    // x = [s, E[y], c] in bf16
    build_x_kernel<<<(BB * KX + 255) / 256, 256, 0, stream>>>(sbuf, embed, ybuf, cbuf, xbf);
    // fused gate GEMM: G = x @ [Wt | Wz | Wr | Wu] + gbias   (M=32, K=2048, N=2560)
    wmma_gemm_m32<<<((NG / 16) * 32 + 255) / 256, 256, 0, stream>>>(
        xbf, KX, WgatesP, gbias, G, (unsigned long long)NG, NG / 16);
    // z, r, r*s, maxout(t_tilde)
    phase2_kernel<<<(BB * DS + 255) / 256, 256, 0, stream>>>(G, sbuf, rs_bf, t_bf, zbuf);
    // v = ss(r*s)                      (M=32, K=512, N=512)
    wmma_gemm_m32<<<(32 * 32 + 255) / 256, 256, 0, stream>>>(
        rs_bf, DS, WssP, ssb, vbuf, (unsigned long long)DS, DS / 16);
    // s = (1-z)s + z tanh(u + v)
    s_update_kernel<<<(BB * DS + 255) / 256, 256, 0, stream>>>(G, vbuf, zbuf, sbuf, sbf);
    // logits = out(t) -> d_out[b][t][:]   (M=32, K=512, N=32000) -- dominant GEMM
    wmma_gemm_m32<<<((VOCAB / 16) * 32 + 255) / 256, 256, 0, stream>>>(
        t_bf, DS, WoutP, outb, out + (size_t)t * VOCAB,
        (unsigned long long)TGT_LEN * VOCAB, VOCAB / 16);
    // greedy feedback
    argmax_kernel<<<BB, 256, 0, stream>>>(out + (size_t)t * VOCAB,
                                          (unsigned long long)TGT_LEN * VOCAB, ybuf);
  }
}
